// MoeBlock_53678501266181
// MI455X (gfx1250) — matrix operand order repaired and checked by reading, not compiled, not run on hardware
//
#include <hip/hip_runtime.h>
#include <hip/hip_bf16.h>
#include <math.h>

typedef __attribute__((ext_vector_type(16))) _Float16 v16h;
typedef __attribute__((ext_vector_type(8)))  _Float16 v8h;
typedef __attribute__((ext_vector_type(8)))  float    v8f;

// ---------------- sizes ----------------
#define Bq   8
#define Sq   512
#define Dq   768
#define Hq   12
#define Eq   14
#define HIDq 1536
#define SHq  3072
#define Mq   (Bq * Sq)        // 4096 rows
#define QKVq (3 * Dq)         // 2304

__device__ __forceinline__ float gelu_exact(float v) {
    return 0.5f * v * (1.0f + erff(v * 0.70710678118654752440f));
}

// ---- CDNA5 async global->LDS copy (16B per lane), tracked by ASYNCcnt ----
__device__ __forceinline__ void async_b128(const _Float16* g, const _Float16* lds) {
    unsigned loff = (unsigned)(unsigned long long)lds;   // generic LDS ptr: low 32b = LDS offset
    asm volatile("global_load_async_to_lds_b128 %0, %1, off"
                 :: "v"(loff), "v"(g)
                 : "memory");
}
template<int N> __device__ __forceinline__ void wait_async() {
    asm volatile("s_wait_asynccnt %0" :: "n"(N) : "memory");
}

// ---------------- elementwise helpers ----------------
__global__ __launch_bounds__(256) void copy_f32(const float* __restrict__ in,
                                                float* __restrict__ out, long n) {
    long i = (long)blockIdx.x * 256 + threadIdx.x;
    if (i < n) out[i] = in[i];
}

// transpose-convert: in f32 [K,N] (batched) -> out f16 [N,K]
__global__ __launch_bounds__(256) void cvtT_kernel(const float* __restrict__ in,
                                                   _Float16* __restrict__ out,
                                                   int K, int N, long sIn, long sOut) {
    __shared__ _Float16 t[32][33];
    const int z = blockIdx.z;
    in  += (long)z * sIn;
    out += (long)z * sOut;
    const int n0 = blockIdx.x * 32, k0 = blockIdx.y * 32;
    const int tx = threadIdx.x & 31, ty = threadIdx.x >> 5;   // 32 x 8
#pragma unroll
    for (int i = 0; i < 4; ++i)
        t[ty + i * 8][tx] = (_Float16)in[(long)(k0 + ty + i * 8) * N + n0 + tx];
    __syncthreads();
#pragma unroll
    for (int i = 0; i < 4; ++i)
        out[(long)(n0 + ty + i * 8) * K + k0 + tx] = t[tx][ty + i * 8];
}

// V^T per (b,h): Vt[z][d][s] = qkv[(b*S+s)*2304 + 1536 + h*64 + d]
__global__ __launch_bounds__(256) void vtrans_kernel(const _Float16* __restrict__ qkv,
                                                     _Float16* __restrict__ vt) {
    __shared__ _Float16 t[32][33];
    const int z = blockIdx.z, b = z / Hq, h = z % Hq;
    const _Float16* in = qkv + (long)b * Sq * QKVq + 2 * Dq + (long)h * 64;
    _Float16* out = vt + (long)z * 64 * Sq;
    const int d0 = blockIdx.x * 32, s0 = blockIdx.y * 32;
    const int tx = threadIdx.x & 31, ty = threadIdx.x >> 5;
#pragma unroll
    for (int i = 0; i < 4; ++i)
        t[ty + i * 8][tx] = in[(long)(s0 + ty + i * 8) * QKVq + d0 + tx];
    __syncthreads();
#pragma unroll
    for (int i = 0; i < 4; ++i)
        out[(long)(d0 + ty + i * 8) * Sq + s0 + tx] = t[tx][ty + i * 8];
}

// ---------------- LayerNorm (768 wide, f16 out) ----------------
__global__ __launch_bounds__(256) void ln_kernel(const float* __restrict__ x,
                                                 const float* __restrict__ g,
                                                 const float* __restrict__ b,
                                                 _Float16* __restrict__ out) {
    __shared__ float red[256];
    const int row = blockIdx.x, tid = threadIdx.x;
    const float* xr = x + (long)row * Dq;
    float v0 = xr[tid], v1 = xr[tid + 256], v2 = xr[tid + 512];
    red[tid] = v0 + v1 + v2;
    __syncthreads();
    for (int s = 128; s > 0; s >>= 1) { if (tid < s) red[tid] += red[tid + s]; __syncthreads(); }
    float mean = red[0] * (1.0f / Dq);
    __syncthreads();
    float d0 = v0 - mean, d1 = v1 - mean, d2 = v2 - mean;
    red[tid] = d0 * d0 + d1 * d1 + d2 * d2;
    __syncthreads();
    for (int s = 128; s > 0; s >>= 1) { if (tid < s) red[tid] += red[tid + s]; __syncthreads(); }
    float rstd = rsqrtf(red[0] * (1.0f / Dq) + 1e-5f);
    _Float16* o = out + (long)row * Dq;
    o[tid]       = (_Float16)(d0 * rstd * g[tid]       + b[tid]);
    o[tid + 256] = (_Float16)(d1 * rstd * g[tid + 256] + b[tid + 256]);
    o[tid + 512] = (_Float16)(d2 * rstd * g[tid + 512] + b[tid + 512]);
}

// ---------------- seq mean over S ----------------
__global__ __launch_bounds__(256) void mean_kernel(const _Float16* __restrict__ h,
                                                   float* __restrict__ seq) {
    int i = blockIdx.x * 256 + threadIdx.x;
    if (i >= Bq * Dq) return;
    int b = i / Dq, d = i % Dq;
    float s = 0.f;
    const _Float16* p = h + (long)b * Sq * Dq + d;
    for (int t = 0; t < Sq; ++t) s += (float)p[(long)t * Dq];
    seq[i] = s * (1.0f / Sq);
}

// ---------------- router dense1: gelu(seq@W1+b1) ----------------
__global__ __launch_bounds__(256) void router1_kernel(const float* __restrict__ seq,
                                                      const float* __restrict__ w1,
                                                      const float* __restrict__ b1,
                                                      float* __restrict__ r1) {
    int i = blockIdx.x * 256 + threadIdx.x;
    if (i >= Bq * Dq) return;
    int b = i / Dq, j = i % Dq;
    float s = b1[j];
    const float* sr = seq + (long)b * Dq;
    for (int k = 0; k < Dq; ++k) s += sr[k] * w1[(long)k * Dq + j];
    r1[i] = gelu_exact(s);
}

// ---------------- router logits + softmax + top4 renorm ----------------
__global__ __launch_bounds__(128) void router2_kernel(const float* __restrict__ r1,
                                                      const float* __restrict__ w2,
                                                      const float* __restrict__ b2,
                                                      float* __restrict__ wout) {
    __shared__ float lg[Bq][Eq];
    int tid = threadIdx.x;
    if (tid < Bq * Eq) {
        int b = tid / Eq, e = tid % Eq;
        float s = b2[e];
        const float* rr = r1 + (long)b * Dq;
        for (int k = 0; k < Dq; ++k) s += rr[k] * w2[(long)k * Eq + e];
        lg[b][e] = s;
    }
    __syncthreads();
    if (tid < Bq) {
        int b = tid;
        float mx = lg[b][0];
        for (int e = 1; e < Eq; ++e) mx = fmaxf(mx, lg[b][e]);
        float pr[Eq]; float sum = 0.f;
        for (int e = 0; e < Eq; ++e) { pr[e] = expf(lg[b][e] - mx); sum += pr[e]; }
        float inv = 1.0f / sum;
        for (int e = 0; e < Eq; ++e) pr[e] *= inv;
        int   idx[4]; float val[4]; bool used[Eq];
        for (int e = 0; e < Eq; ++e) used[e] = false;
        for (int t = 0; t < 4; ++t) {
            int bi = 0; float bv = -1.f;
            for (int e = 0; e < Eq; ++e)
                if (!used[e] && pr[e] > bv) { bv = pr[e]; bi = e; }
            used[bi] = true; idx[t] = bi; val[t] = bv;
        }
        float m2 = val[0], s2 = 0.f, tw[4];
        for (int t = 0; t < 4; ++t) { tw[t] = expf(val[t] - m2); s2 += tw[t]; }
        float inv2 = 1.0f / s2;
        for (int e = 0; e < Eq; ++e) wout[b * Eq + e] = 0.f;
        for (int t = 0; t < 4; ++t) wout[b * Eq + idx[t]] = tw[t] * inv2;
    }
}

// ---------------- softmax over rows of 512 (scores f32 -> P f16) ----------------
__global__ __launch_bounds__(256) void softmax_kernel(const float* __restrict__ sc,
                                                      _Float16* __restrict__ P) {
    __shared__ float red[256];
    long row = blockIdx.x;
    const float* s = sc + row * Sq;
    int tid = threadIdx.x;
    float a = s[tid], b = s[tid + 256];
    red[tid] = fmaxf(a, b);
    __syncthreads();
    for (int t = 128; t > 0; t >>= 1) { if (tid < t) red[tid] = fmaxf(red[tid], red[tid + t]); __syncthreads(); }
    float mx = red[0];
    __syncthreads();
    float ea = expf(a - mx), eb = expf(b - mx);
    red[tid] = ea + eb;
    __syncthreads();
    for (int t = 128; t > 0; t >>= 1) { if (tid < t) red[tid] += red[tid + t]; __syncthreads(); }
    float inv = 1.0f / red[0];
    P[row * Sq + tid]       = (_Float16)(ea * inv);
    P[row * Sq + tid + 256] = (_Float16)(eb * inv);
}

// ---------------- generic batched WMMA GEMM, async double-buffered ----------------
// 256 thr = 8 wave32; block tile 128(M) x 64(N); K-tile 64; wave tile 32x32 = 2x2 WMMA.
// A: f16 [M,K] row-major.  B: f16 [N,K] row-major (pre-transposed).
// epilogue: v = alpha*acc + bias[n]; if GELU v=gelu(v); v *= rowscale[(row/512)*stride];
//           if ACCUM v += C; store C (f32, if C) and Ch (f16, if Ch).
template<bool DOGELU, bool ACCUM>
__global__ __launch_bounds__(256) void wmma_gemm(
    const _Float16* __restrict__ Ap, int lda, long sAb, long sAh,
    const _Float16* __restrict__ Bp, int ldb, long sBb, long sBh,
    float* __restrict__ Cp, int ldc, long sCb, long sCh,
    _Float16* __restrict__ Chp, int ldch, long sChb, long sChh,
    const float* __restrict__ bias,
    const float* __restrict__ rowscale, int rsStride,
    float alpha, int K, int batchH)
{
    __shared__ alignas(32) _Float16 As[2][128][64];   // [buf][m][k]  32 KB
    __shared__ alignas(32) _Float16 Bs[2][64][64];    // [buf][n][k]  16 KB

    const int z  = blockIdx.z;
    const int zb = z / batchH, zh = z % batchH;
    const _Float16* A = Ap + zb * sAb + zh * sAh;
    const _Float16* B = Bp + zb * sBb + zh * sBh;
    float*    C  = Cp  ? Cp  + zb * sCb  + zh * sCh  : nullptr;
    _Float16* Ch = Chp ? Chp + zb * sChb + zh * sChh : nullptr;

    const int m0 = blockIdx.y * 128;
    const int n0 = blockIdx.x * 64;
    const int tid  = threadIdx.x;
    const int lane = tid & 31, wave = tid >> 5;
    const int wm = wave & 3, wn = wave >> 2;          // 4x2 wave grid
    const int r15 = lane & 15, hi16 = lane >> 4;

    // issue one 64-deep K tile: A = 1024 x 16B chunks, B = 512 x 16B chunks
    auto issue = [&](int kt, int buf) {
        const int tK = kt * 64;
#pragma unroll
        for (int i = 0; i < 4; ++i) {                 // A: 4 chunks/thread
            int c = tid + i * 256;
            int row = c >> 3, kh = (c & 7) * 8;
            async_b128(A + (long)(m0 + row) * lda + tK + kh, &As[buf][row][kh]);
        }
#pragma unroll
        for (int i = 0; i < 2; ++i) {                 // B: 2 chunks/thread
            int c = tid + i * 256;
            int n = c >> 3, kh = (c & 7) * 8;
            async_b128(B + (long)(n0 + n) * ldb + tK + kh, &Bs[buf][n][kh]);
        }
    };

    v8f acc[2][2] = {};
    const int T = K >> 6;                             // K / 64 tiles
    issue(0, 0);
    for (int t = 0; t < T; ++t) {
        const int cur = t & 1;
        if (t + 1 < T) { issue(t + 1, cur ^ 1); wait_async<6>(); }
        else          { wait_async<0>(); }
        __syncthreads();
#pragma unroll
        for (int ks = 0; ks < 64; ks += 32) {
            v16h af[2], bf[2];
#pragma unroll
            for (int i = 0; i < 2; ++i) {
                const _Float16* pa = &As[cur][wm * 32 + i * 16 + r15][ks + hi16 * 16];
                af[i] = *(const v16h*)pa;
            }
#pragma unroll
            for (int j = 0; j < 2; ++j) {
                const _Float16* pb = &Bs[cur][wn * 32 + j * 16 + r15][ks + hi16 * 16];
                bf[j] = *(const v16h*)pb;
            }
#pragma unroll
            for (int i = 0; i < 2; ++i)
#pragma unroll
                for (int j = 0; j < 2; ++j)
                    acc[i][j] = __builtin_amdgcn_wmma_f32_16x16x32_f16(
                        false, af[i], false, bf[j], (short)0, acc[i][j], false, false);
        }
        __syncthreads();
    }

    // ---- epilogue ----
    const int rbase = hi16 * 8;
#pragma unroll
    for (int i = 0; i < 2; ++i) {
#pragma unroll
        for (int j = 0; j < 2; ++j) {
            const int col = n0 + wn * 32 + j * 16 + r15;
            const float bv = bias ? bias[col] : 0.0f;
#pragma unroll
            for (int r = 0; r < 8; ++r) {
                const int row = m0 + wm * 32 + i * 16 + rbase + r;
                float v = alpha * acc[i][j][r] + bv;
                if (DOGELU) v = gelu_exact(v);
                if (rowscale) v *= rowscale[(row >> 9) * rsStride];  // 512 rows/batch
                if (C) {
                    long ci = (long)row * ldc + col;
                    if (ACCUM) v += C[ci];
                    C[ci] = v;
                }
                if (Ch) Ch[(long)row * ldch + col] = (_Float16)v;
            }
        }
    }
}

// ---------------- launcher ----------------
extern "C" void kernel_launch(void* const* d_in, const int* in_sizes, int n_in,
                              void* d_out, int out_size, void* d_ws, size_t ws_size,
                              hipStream_t stream) {
    const float* x      = (const float*)d_in[0];
    const float* ln1_g  = (const float*)d_in[1];
    const float* ln1_b  = (const float*)d_in[2];
    const float* qkv_w  = (const float*)d_in[3];
    const float* proj_w = (const float*)d_in[4];
    const float* proj_b = (const float*)d_in[5];
    const float* ln2_g  = (const float*)d_in[6];
    const float* ln2_b  = (const float*)d_in[7];
    const float* rw1    = (const float*)d_in[8];
    const float* rb1    = (const float*)d_in[9];
    const float* rw2    = (const float*)d_in[10];
    const float* rb2    = (const float*)d_in[11];
    const float* exp_w1 = (const float*)d_in[12];
    const float* exp_b1 = (const float*)d_in[13];
    const float* exp_w2 = (const float*)d_in[14];
    const float* exp_b2 = (const float*)d_in[15];
    const float* sh_w1  = (const float*)d_in[16];
    const float* sh_b1  = (const float*)d_in[17];
    const float* sh_w2  = (const float*)d_in[18];
    const float* sh_b2  = (const float*)d_in[19];
    float* out = (float*)d_out;

    char* base = (char*)d_ws;
    size_t off = 0;
    auto take = [&](size_t bytes) -> void* {
        void* p = base + off;
        off += (bytes + 255) & ~(size_t)255;
        return p;
    };
    // transposed f16 weights [N,K]
    _Float16* qkvWt = (_Float16*)take((size_t)QKVq * Dq * 2);
    _Float16* projWt= (_Float16*)take((size_t)Dq * Dq * 2);
    _Float16* ew1t  = (_Float16*)take((size_t)Eq * HIDq * Dq * 2);
    _Float16* ew2t  = (_Float16*)take((size_t)Eq * Dq * HIDq * 2);
    _Float16* sw1t  = (_Float16*)take((size_t)SHq * Dq * 2);
    _Float16* sw2t  = (_Float16*)take((size_t)Dq * SHq * 2);
    // activations
    _Float16* h1h   = (_Float16*)take((size_t)Mq * Dq * 2);
    _Float16* qkvh  = (_Float16*)take((size_t)Mq * QKVq * 2);
    _Float16* Vt    = (_Float16*)take((size_t)Bq * Hq * 64 * Sq * 2);
    float*    scores= (float*)   take((size_t)Bq * Hq * Sq * Sq * 4);
    _Float16* Pp    = (_Float16*)take((size_t)Bq * Hq * Sq * Sq * 2);
    _Float16* aoh   = (_Float16*)take((size_t)Mq * Dq * 2);
    float*    x2    = (float*)   take((size_t)Mq * Dq * 4);
    _Float16* hh    = (_Float16*)take((size_t)Mq * Dq * 2);
    float*    seq   = (float*)   take((size_t)Bq * Dq * 4);
    float*    r1    = (float*)   take((size_t)Bq * Dq * 4);
    float*    wgt   = (float*)   take((size_t)Bq * Eq * 4);
    _Float16* hidh  = (_Float16*)take((size_t)Mq * HIDq * 2);
    _Float16* shidh = (_Float16*)take((size_t)Mq * SHq * 2);

    // 1. transpose-convert all weights to f16 [N,K]
    cvtT_kernel<<<dim3(QKVq/32, Dq/32, 1), 256, 0, stream>>>(qkv_w, qkvWt, Dq, QKVq, 0, 0);
    cvtT_kernel<<<dim3(Dq/32, Dq/32, 1), 256, 0, stream>>>(proj_w, projWt, Dq, Dq, 0, 0);
    cvtT_kernel<<<dim3(HIDq/32, Dq/32, Eq), 256, 0, stream>>>(exp_w1, ew1t, Dq, HIDq,
                                                              (long)Dq*HIDq, (long)HIDq*Dq);
    cvtT_kernel<<<dim3(Dq/32, HIDq/32, Eq), 256, 0, stream>>>(exp_w2, ew2t, HIDq, Dq,
                                                              (long)HIDq*Dq, (long)Dq*HIDq);
    cvtT_kernel<<<dim3(SHq/32, Dq/32, 1), 256, 0, stream>>>(sh_w1, sw1t, Dq, SHq, 0, 0);
    cvtT_kernel<<<dim3(Dq/32, SHq/32, 1), 256, 0, stream>>>(sh_w2, sw2t, SHq, Dq, 0, 0);

    // 2. LN1
    ln_kernel<<<Mq, 256, 0, stream>>>(x, ln1_g, ln1_b, h1h);

    // 3. QKV GEMM -> f16
    wmma_gemm<false,false><<<dim3(QKVq/64, Mq/128, 1), 256, 0, stream>>>(
        h1h, Dq, 0, 0,  qkvWt, Dq, 0, 0,
        nullptr, 0, 0, 0,  qkvh, QKVq, 0, 0,
        nullptr, nullptr, 0, 1.0f, Dq, 1);

    // 4. V^T per head
    vtrans_kernel<<<dim3(2, Sq/32, Bq*Hq), 256, 0, stream>>>(qkvh, Vt);

    // 5. scores = Q K^T * dh^-0.5   (batched over 96 = B*H)
    wmma_gemm<false,false><<<dim3(Sq/64, Sq/128, Bq*Hq), 256, 0, stream>>>(
        qkvh, QKVq, (long)Sq*QKVq, 64,
        qkvh + Dq, QKVq, (long)Sq*QKVq, 64,
        scores, Sq, (long)Hq*Sq*Sq, (long)Sq*Sq,
        nullptr, 0, 0, 0,
        nullptr, nullptr, 0, 0.125f, 64, Hq);

    // 6. softmax rows -> P (f16)
    softmax_kernel<<<Bq*Hq*Sq, 256, 0, stream>>>(scores, Pp);

    // 7. P @ V -> attn out f16 in [B,S,H*dh] layout
    wmma_gemm<false,false><<<dim3(1, Sq/128, Bq*Hq), 256, 0, stream>>>(
        Pp, Sq, (long)Hq*Sq*Sq, (long)Sq*Sq,
        Vt, Sq, (long)Hq*64*Sq, (long)64*Sq,
        nullptr, 0, 0, 0,
        aoh, Dq, (long)Sq*Dq, 64,
        nullptr, nullptr, 0, 1.0f, Sq, Hq);

    // 8. x2 = x + proj(ao)
    copy_f32<<<(Mq*Dq + 255)/256, 256, 0, stream>>>(x, x2, (long)Mq*Dq);
    wmma_gemm<false,true><<<dim3(Dq/64, Mq/128, 1), 256, 0, stream>>>(
        aoh, Dq, 0, 0,  projWt, Dq, 0, 0,
        x2, Dq, 0, 0,  nullptr, 0, 0, 0,
        proj_b, nullptr, 0, 1.0f, Dq, 1);

    // 9. LN2 + router
    ln_kernel<<<Mq, 256, 0, stream>>>(x2, ln2_g, ln2_b, hh);
    mean_kernel<<<(Bq*Dq + 255)/256, 256, 0, stream>>>(hh, seq);
    router1_kernel<<<(Bq*Dq + 255)/256, 256, 0, stream>>>(seq, rw1, rb1, r1);
    router2_kernel<<<1, 128, 0, stream>>>(r1, rw2, rb2, wgt);

    // 10. out = x2, then accumulate MoE + shared
    copy_f32<<<(Mq*Dq + 255)/256, 256, 0, stream>>>(x2, out, (long)Mq*Dq);

    for (int e = 0; e < Eq; ++e) {
        wmma_gemm<true,false><<<dim3(HIDq/64, Mq/128, 1), 256, 0, stream>>>(
            hh, Dq, 0, 0,
            ew1t + (long)e*HIDq*Dq, Dq, 0, 0,
            nullptr, 0, 0, 0,  hidh, HIDq, 0, 0,
            exp_b1 + (long)e*HIDq, nullptr, 0, 1.0f, Dq, 1);
        wmma_gemm<false,true><<<dim3(Dq/64, Mq/128, 1), 256, 0, stream>>>(
            hidh, HIDq, 0, 0,
            ew2t + (long)e*Dq*HIDq, HIDq, 0, 0,
            out, Dq, 0, 0,  nullptr, 0, 0, 0,
            exp_b2 + (long)e*Dq, wgt + e, Eq, 1.0f, HIDq, 1);
    }

    // 11. shared expert
    wmma_gemm<true,false><<<dim3(SHq/64, Mq/128, 1), 256, 0, stream>>>(
        hh, Dq, 0, 0,  sw1t, Dq, 0, 0,
        nullptr, 0, 0, 0,  shidh, SHq, 0, 0,
        sh_b1, nullptr, 0, 1.0f, Dq, 1);
    wmma_gemm<false,true><<<dim3(Dq/64, Mq/128, 1), 256, 0, stream>>>(
        shidh, SHq, 0, 0,  sw2t, SHq, 0, 0,
        out, Dq, 0, 0,  nullptr, 0, 0, 0,
        sh_b2, nullptr, 0, 1.0f, SHq, 1);

    (void)in_sizes; (void)n_in; (void)out_size; (void)ws_size;
}
